// SelfAttention_16896401343214
// MI455X (gfx1250) — compile-verified, matrix operand order checked by reading, not run on hardware
//
#include <hip/hip_runtime.h>
#include <hip/hip_bf16.h>

typedef __attribute__((ext_vector_type(16))) _Float16 v16h;
typedef __attribute__((ext_vector_type(8)))  _Float16 v8h;
typedef __attribute__((ext_vector_type(8)))  float    v8f;
typedef __attribute__((ext_vector_type(4)))  float    v4f;

#define DIMS 512
#define NB   16
#define HW   4096   // 64*64
#define HW4  1024   // pooled positions
#define C8   64
#define C2   256

#define WMMA_F16(A, B, C) __builtin_amdgcn_wmma_f32_16x16x32_f16(false, A, false, B, (short)0, C, false, false)

// Fragment K pattern for 16x16x32 f16 WMMA (ISA 7.12.2):
//   lane half 0: K in [0,8) and [16,24);  lane half 1: K in [8,16) and [24,32).
// With K contiguous in memory a fragment is exactly two 16-byte loads.
__device__ __forceinline__ v16h load_frag_row(const _Float16* __restrict__ p, int half) {
    v8h lo = *(const v8h*)(p + half * 8);
    v8h hi = *(const v8h*)(p + 16 + half * 8);
    v16h r;
    #pragma unroll
    for (int i = 0; i < 8; ++i) { r[i] = lo[i]; r[i + 8] = hi[i]; }
    return r;
}

// ---------------------------------------------------------------------------
// Fused QKV 1x1 conv: reads X once, produces theta, phi-conv and g-conv.
// Per wave: 6 accumulators = theta tile (16 co) + phi tile (16 co) + 4 g tiles
// (64 co). Block of 4 waves covers all 64+64+256 output channels.
// Grid (HW/16, 1, B), block 128.
// ---------------------------------------------------------------------------
__global__ void qkv_conv_kernel(const float* __restrict__ X,
                                const _Float16* __restrict__ Wth,
                                const _Float16* __restrict__ Wph,
                                const _Float16* __restrict__ Wg,
                                const float* __restrict__ bth,
                                const float* __restrict__ bph,
                                const float* __restrict__ bg,
                                _Float16* __restrict__ theta,
                                _Float16* __restrict__ phic,
                                _Float16* __restrict__ gc) {
    __shared__ _Float16 sA[16 * 128];   // [m][ci_panel], K contiguous per row

    const int hw0  = blockIdx.x * 16;
    const int b    = blockIdx.z;
    const int t    = threadIdx.x;
    const int lane = t & 31;
    const int wave = t >> 5;
    const int half = lane >> 4;
    const int mn   = lane & 15;
    const int m_s  = t & 15;            // staging row
    const int ci0  = (t >> 4) * 16;     // staging K base

    const float* xb = X + (size_t)b * DIMS * HW + hw0;
    const _Float16* wrT  = Wth + (size_t)(wave * 16 + mn) * DIMS;
    const _Float16* wrP  = Wph + (size_t)(wave * 16 + mn) * DIMS;
    const _Float16* wrG0 = Wg + (size_t)(wave * 64 +      mn) * DIMS;
    const _Float16* wrG1 = Wg + (size_t)(wave * 64 + 16 + mn) * DIMS;
    const _Float16* wrG2 = Wg + (size_t)(wave * 64 + 32 + mn) * DIMS;
    const _Float16* wrG3 = Wg + (size_t)(wave * 64 + 48 + mn) * DIMS;

    v8f accT = {}, accP = {}, accG0 = {}, accG1 = {}, accG2 = {}, accG3 = {};

    for (int kp = 0; kp < DIMS; kp += 128) {
        __syncthreads();
        // gather 16 strided f32 (independent loads -> one clause), pack, 2x b128
        float tmp[16];
        #pragma unroll
        for (int j = 0; j < 16; ++j)
            tmp[j] = xb[(size_t)(kp + ci0 + j) * HW + m_s];
        v8h h0, h1;
        #pragma unroll
        for (int j = 0; j < 8; ++j) {
            h0[j] = (_Float16)tmp[j];
            h1[j] = (_Float16)tmp[j + 8];
        }
        *(v8h*)(sA + m_s * 128 + ci0)     = h0;
        *(v8h*)(sA + m_s * 128 + ci0 + 8) = h1;
        if (kp + 128 < DIMS)              // prefetch next panel
            __builtin_prefetch(&xb[(size_t)(kp + 128 + ci0) * HW + m_s], 0, 0);
        __syncthreads();

        #pragma unroll
        for (int q = 0; q < 4; ++q) {
            const int kk = q * 32;
            v16h a  = load_frag_row(sA + mn * 128 + kk, half);
            v16h bT = load_frag_row(wrT  + kp + kk, half);
            v16h bP = load_frag_row(wrP  + kp + kk, half);
            v16h b0 = load_frag_row(wrG0 + kp + kk, half);
            v16h b1 = load_frag_row(wrG1 + kp + kk, half);
            v16h b2 = load_frag_row(wrG2 + kp + kk, half);
            v16h b3 = load_frag_row(wrG3 + kp + kk, half);
            accT  = WMMA_F16(a, bT, accT);
            accP  = WMMA_F16(a, bP, accP);
            accG0 = WMMA_F16(a, b0, accG0);
            accG1 = WMMA_F16(a, b1, accG1);
            accG2 = WMMA_F16(a, b2, accG2);
            accG3 = WMMA_F16(a, b3, accG3);
        }
    }

    const float bvT  = bth[wave * 16 + mn];
    const float bvP  = bph[wave * 16 + mn];
    const float bvG0 = bg[wave * 64 +      mn];
    const float bvG1 = bg[wave * 64 + 16 + mn];
    const float bvG2 = bg[wave * 64 + 32 + mn];
    const float bvG3 = bg[wave * 64 + 48 + mn];
    #pragma unroll
    for (int r = 0; r < 8; ++r) {
        int m = r + half * 8;
        size_t row = (size_t)b * HW + hw0 + m;
        theta[row * C8 + wave * 16 + mn] = (_Float16)(accT[r] + bvT);
        phic [row * C8 + wave * 16 + mn] = (_Float16)(accP[r] + bvP);
        gc[row * C2 + wave * 64 +      mn] = (_Float16)(accG0[r] + bvG0);
        gc[row * C2 + wave * 64 + 16 + mn] = (_Float16)(accG1[r] + bvG1);
        gc[row * C2 + wave * 64 + 32 + mn] = (_Float16)(accG2[r] + bvG2);
        gc[row * C2 + wave * 64 + 48 + mn] = (_Float16)(accG3[r] + bvG3);
    }
}

// ---------------------------------------------------------------------------
// 2x2 maxpool, phi: [b][hw][64] -> phi[b][key][64]; v8h in/out.
// ---------------------------------------------------------------------------
__global__ void pool_phi_kernel(const _Float16* __restrict__ pc,
                                _Float16* __restrict__ phi) {
    int idx = blockIdx.x * blockDim.x + threadIdx.x;   // NB*HW4*(C8/8)
    if (idx >= NB * HW4 * (C8 / 8)) return;
    int cb = idx & 7;
    int mp = (idx >> 3) & (HW4 - 1);
    int b  = idx >> 13;
    int hp = mp >> 5, wp = mp & 31;
    int hw = (hp * 2) * 64 + wp * 2;
    const _Float16* base = pc + (size_t)b * HW * C8 + cb * 8;
    v8h v0 = *(const v8h*)(base + (size_t)(hw)      * C8);
    v8h v1 = *(const v8h*)(base + (size_t)(hw + 1)  * C8);
    v8h v2 = *(const v8h*)(base + (size_t)(hw + 64) * C8);
    v8h v3 = *(const v8h*)(base + (size_t)(hw + 65) * C8);
    v8h m;
    #pragma unroll
    for (int i = 0; i < 8; ++i)
        m[i] = (_Float16)fmaxf(fmaxf((float)v0[i], (float)v1[i]),
                               fmaxf((float)v2[i], (float)v3[i]));
    *(v8h*)(phi + ((size_t)b * HW4 + mp) * C8 + cb * 8) = m;
}

// 2x2 maxpool, g: [b][hw][256] -> g[b][c][key]; 8 consecutive keys per thread.
__global__ void pool_g_kernel(const _Float16* __restrict__ gc,
                              _Float16* __restrict__ g) {
    int idx = blockIdx.x * blockDim.x + threadIdx.x;   // NB*C2*(HW4/8)
    if (idx >= NB * C2 * (HW4 / 8)) return;
    int kb = idx & 127;
    int c  = (idx >> 7) & (C2 - 1);
    int b  = idx >> 15;
    int mp0 = kb * 8;
    int hp = mp0 >> 5, wp0 = mp0 & 31;
    const _Float16* base = gc + (size_t)b * HW * C2 + c;
    v8h m;
    #pragma unroll
    for (int i = 0; i < 8; ++i) {
        int hw = (hp * 2) * 64 + (wp0 + i) * 2;
        float v0 = (float)base[(size_t)(hw)      * C2];
        float v1 = (float)base[(size_t)(hw + 1)  * C2];
        float v2 = (float)base[(size_t)(hw + 64) * C2];
        float v3 = (float)base[(size_t)(hw + 65) * C2];
        m[i] = (_Float16)fmaxf(fmaxf(v0, v1), fmaxf(v2, v3));
    }
    *(v8h*)(g + ((size_t)b * C2 + c) * HW4 + mp0) = m;
}

// f32 -> f16 conversion (weights)
__global__ void cvt_f16_kernel(const float* __restrict__ src,
                               _Float16* __restrict__ dst, int n) {
    int i = blockIdx.x * blockDim.x + threadIdx.x;
    if (i * 4 >= n) return;
    v4f v = *(const v4f*)(src + i * 4);
    _Float16* d = dst + i * 4;
    d[0] = (_Float16)v[0]; d[1] = (_Float16)v[1];
    d[2] = (_Float16)v[2]; d[3] = (_Float16)v[3];
}

// ---------------------------------------------------------------------------
// Fused attention per 16-row query tile:
//   S(16x1024)=theta@phi (WMMA) -> LDS f32 softmax -> O=P@g (WMMA, A reused 4x)
// Block 128 = 4 waves. Grid (HW/16, B). Dynamic LDS = 100,864 B (<320 KB/WGP).
// ---------------------------------------------------------------------------
__global__ void attn_wmma_kernel(const _Float16* __restrict__ theta,
                                 const _Float16* __restrict__ phi,
                                 const _Float16* __restrict__ g,
                                 _Float16* __restrict__ o) {
    extern __shared__ char smem[];
    _Float16* sTheta = (_Float16*)smem;                        // 16x64   (2 KB)
    float*    sS     = (float*)(smem + 2048);                  // 16x1024 (64 KB)
    _Float16* sP     = (_Float16*)(smem + 2048 + 65536);       // 16x1024 (32 KB)
    float*    sRed   = (float*)(smem + 2048 + 65536 + 32768);  // 16x8

    const int hw0  = blockIdx.x * 16;
    const int b    = blockIdx.y;
    const int t    = threadIdx.x;
    const int lane = t & 31;
    const int wave = t >> 5;
    const int half = lane >> 4;
    const int mn   = lane & 15;

    // theta tile = 1024 contiguous halfwords: one v8h per thread
    ((v8h*)sTheta)[t] = ((const v8h*)(theta + ((size_t)b * HW + hw0) * C8))[t];
    __syncthreads();

    // ---- Phase 1: scores; B-fragments straight from key-major phi ----
    for (int kt = 0; kt < 16; ++kt) {
        int key0 = wave * 256 + kt * 16;
        const _Float16* prow = phi + ((size_t)b * HW4 + key0 + mn) * C8;
        v16h a0 = load_frag_row(sTheta + mn * C8, half);
        v16h a1 = load_frag_row(sTheta + mn * C8 + 32, half);
        v16h b0 = load_frag_row(prow, half);
        v16h b1 = load_frag_row(prow + 32, half);
        v8f acc = {};
        acc = WMMA_F16(a0, b0, acc);
        acc = WMMA_F16(a1, b1, acc);
        #pragma unroll
        for (int r = 0; r < 8; ++r)
            sS[(r + half * 8) * HW4 + key0 + mn] = acc[r];
    }
    __syncthreads();

    // ---- Phase 2: softmax over 1024 keys, 8 threads per row, vectorized ----
    {
        int row = t >> 3, seg = t & 7;
        const v4f* s4 = (const v4f*)(sS + row * HW4 + seg * 128);   // 32 v4f
        v8h*       p8 = (v8h*)(sP + row * HW4 + seg * 128);         // 16 v8h

        float lmax = -3.4e38f;
        #pragma unroll 8
        for (int i = 0; i < 32; ++i) {
            v4f v = s4[i];
            lmax = fmaxf(lmax, fmaxf(fmaxf(v[0], v[1]), fmaxf(v[2], v[3])));
        }
        sRed[row * 8 + seg] = lmax;
        __syncthreads();
        float rmax = -3.4e38f;
        #pragma unroll
        for (int i = 0; i < 8; ++i) rmax = fmaxf(rmax, sRed[row * 8 + i]);
        __syncthreads();

        float lsum = 0.f;
        #pragma unroll 4
        for (int i = 0; i < 16; ++i) {
            v4f a = s4[2 * i], c = s4[2 * i + 1];
            float e[8];
            e[0] = __expf(a[0] - rmax); e[1] = __expf(a[1] - rmax);
            e[2] = __expf(a[2] - rmax); e[3] = __expf(a[3] - rmax);
            e[4] = __expf(c[0] - rmax); e[5] = __expf(c[1] - rmax);
            e[6] = __expf(c[2] - rmax); e[7] = __expf(c[3] - rmax);
            v8h hh;
            #pragma unroll
            for (int j = 0; j < 8; ++j) { lsum += e[j]; hh[j] = (_Float16)e[j]; }
            p8[i] = hh;
        }
        sRed[row * 8 + seg] = lsum;
        __syncthreads();
        float tot = 0.f;
        #pragma unroll
        for (int i = 0; i < 8; ++i) tot += sRed[row * 8 + i];
        float inv = 1.0f / tot;
        #pragma unroll 4
        for (int i = 0; i < 16; ++i) {
            v8h hh = p8[i];
            #pragma unroll
            for (int j = 0; j < 8; ++j) hh[j] = (_Float16)((float)hh[j] * inv);
            p8[i] = hh;
        }
    }
    __syncthreads();

    // ---- Phase 3: O = P @ g ; wave owns 64 channels; g is channel-major ----
    v8f acc0 = {}, acc1 = {}, acc2 = {}, acc3 = {};
    const int c0 = wave * 64;
    const _Float16* g0 = g + ((size_t)b * C2 + c0 +      mn) * HW4;
    const _Float16* g1 = g + ((size_t)b * C2 + c0 + 16 + mn) * HW4;
    const _Float16* g2 = g + ((size_t)b * C2 + c0 + 32 + mn) * HW4;
    const _Float16* g3 = g + ((size_t)b * C2 + c0 + 48 + mn) * HW4;
    for (int kk = 0; kk < HW4; kk += 32) {
        v16h a  = load_frag_row(sP + mn * HW4 + kk, half);
        v16h b0 = load_frag_row(g0 + kk, half);
        v16h b1 = load_frag_row(g1 + kk, half);
        v16h b2 = load_frag_row(g2 + kk, half);
        v16h b3 = load_frag_row(g3 + kk, half);
        acc0 = WMMA_F16(a, b0, acc0);
        acc1 = WMMA_F16(a, b1, acc1);
        acc2 = WMMA_F16(a, b2, acc2);
        acc3 = WMMA_F16(a, b3, acc3);
    }
    #pragma unroll
    for (int r = 0; r < 8; ++r) {
        int m = r + half * 8;
        size_t orow = ((size_t)b * HW + hw0 + m) * C2;
        o[orow + c0 +      mn] = (_Float16)acc0[r];
        o[orow + c0 + 16 + mn] = (_Float16)acc1[r];
        o[orow + c0 + 32 + mn] = (_Float16)acc2[r];
        o[orow + c0 + 48 + mn] = (_Float16)acc3[r];
    }
}

// ---------------------------------------------------------------------------
// Final: out[b][co][hw] = x + sigma*(w_attn[co,:]·attnG[b][hw,:] + b_attn[co])
// Grid (HW/16, 512/64, B), block 128 = 4 waves. Loads hoisted 4-deep.
// ---------------------------------------------------------------------------
__global__ void final_conv_kernel(const _Float16* __restrict__ ag,
                                  const _Float16* __restrict__ Wh,
                                  const float* __restrict__ bt,
                                  const float* __restrict__ x,
                                  const float* __restrict__ sigma,
                                  float* __restrict__ out) {
    const int hw0  = blockIdx.x * 16;
    const int cg   = blockIdx.y;
    const int b    = blockIdx.z;
    const int t    = threadIdx.x;
    const int lane = t & 31;
    const int wave = t >> 5;
    const int half = lane >> 4;
    const int mn   = lane & 15;
    const int co0  = cg * 64 + wave * 16;

    const _Float16* arow = ag + ((size_t)b * HW + hw0 + mn) * C2;
    const _Float16* wrow = Wh + (size_t)(co0 + mn) * C2;
    v8f acc = {};
    #pragma unroll
    for (int kp = 0; kp < C2; kp += 128) {
        v16h a[4], bb[4];
        #pragma unroll
        for (int q = 0; q < 4; ++q) {
            a[q]  = load_frag_row(arow + kp + q * 32, half);
            bb[q] = load_frag_row(wrow + kp + q * 32, half);
        }
        #pragma unroll
        for (int q = 0; q < 4; ++q)
            acc = WMMA_F16(a[q], bb[q], acc);
    }
    const float s  = sigma[0];
    const float bv = bt[co0 + mn];
    #pragma unroll
    for (int r = 0; r < 8; ++r) {
        int m = r + half * 8;
        size_t idx = ((size_t)b * DIMS + co0 + mn) * (size_t)HW + hw0 + m;
        out[idx] = x[idx] + s * (acc[r] + bv);
    }
}

// ---------------------------------------------------------------------------
extern "C" void kernel_launch(void* const* d_in, const int* in_sizes, int n_in,
                              void* d_out, int out_size, void* d_ws, size_t ws_size,
                              hipStream_t stream) {
    const float* x       = (const float*)d_in[0];
    const float* w_theta = (const float*)d_in[1];
    const float* b_theta = (const float*)d_in[2];
    const float* w_phi   = (const float*)d_in[3];
    const float* b_phi   = (const float*)d_in[4];
    const float* w_g     = (const float*)d_in[5];
    const float* b_g     = (const float*)d_in[6];
    const float* w_attn  = (const float*)d_in[7];
    const float* b_attn  = (const float*)d_in[8];
    const float* sigma   = (const float*)d_in[9];
    float* out = (float*)d_out;

    char* ws = (char*)d_ws;
    _Float16* theta   = (_Float16*)(ws);                 //  8 MB   [b][hw][64]
    _Float16* phi     = (_Float16*)(ws +  8388608);      //  2 MB   [b][key][64]
    _Float16* gbuf    = (_Float16*)(ws + 10485760);      //  8 MB   [b][c][key]
    _Float16* wth_h   = (_Float16*)(ws + 18874368);      //  64 KB  [64][512]
    _Float16* wph_h   = (_Float16*)(ws + 18939904);      //  64 KB  [64][512]
    _Float16* wg_h    = (_Float16*)(ws + 19005440);      // 256 KB  [256][512]
    _Float16* wat_h   = (_Float16*)(ws + 19267584);      // 256 KB  [512][256]
    _Float16* phic    = (_Float16*)(ws + 19529728);      //  8 MB   [b][hw][64]
    _Float16* gconv   = (_Float16*)(ws + 27918336);      //  32 MB  [b][hw][256]
    _Float16* attnout = (_Float16*)(ws + 19529728);      //  reuses phic+gconv

    const dim3 blk(128);

    cvt_f16_kernel<<<(C8 * DIMS / 4 + 255) / 256, 256, 0, stream>>>(w_theta, wth_h, C8 * DIMS);
    cvt_f16_kernel<<<(C8 * DIMS / 4 + 255) / 256, 256, 0, stream>>>(w_phi, wph_h, C8 * DIMS);
    cvt_f16_kernel<<<(C2 * DIMS / 4 + 255) / 256, 256, 0, stream>>>(w_g, wg_h, C2 * DIMS);
    cvt_f16_kernel<<<(DIMS * C2 / 4 + 255) / 256, 256, 0, stream>>>(w_attn, wat_h, DIMS * C2);

    // one pass over X produces theta, phi-conv and g-conv
    qkv_conv_kernel<<<dim3(HW / 16, 1, NB), blk, 0, stream>>>(
        x, wth_h, wph_h, wg_h, b_theta, b_phi, b_g, theta, phic, gconv);
    pool_phi_kernel<<<(NB * HW4 * (C8 / 8)) / 256, 256, 0, stream>>>(phic, phi);
    pool_g_kernel<<<(NB * C2 * (HW4 / 8)) / 256, 256, 0, stream>>>(gconv, gbuf);
    attn_wmma_kernel<<<dim3(HW / 16, NB), blk, 100864, stream>>>(
        theta, phi, gbuf, attnout);
    final_conv_kernel<<<dim3(HW / 16, DIMS / 64, NB), blk, 0, stream>>>(
        attnout, wat_h, b_attn, x, sigma, out);
}
